// ViT_MoE_v3_79912161509416
// MI455X (gfx1250) — matrix operand order repaired and checked by reading, not compiled, not run on hardware
//
#include <hip/hip_runtime.h>
#include <hip/hip_bf16.h>
#include <math.h>

// ---------------------------------------------------------------------------
// ViT-MoE forward for MI455X (gfx1250), bf16 WMMA GEMM core.
// B=64, E=256, H=8, NE=4, K(top)=2, P=16, IMG=224, NC=38, FF=1024, DEPTH=4
// S=197, T=B*S=12608
// ---------------------------------------------------------------------------

typedef __bf16 bf16_t;
typedef bf16_t v4bf  __attribute__((ext_vector_type(4)));
typedef bf16_t v8bf  __attribute__((ext_vector_type(8)));
typedef bf16_t v16bf __attribute__((ext_vector_type(16)));
typedef float  v4f   __attribute__((ext_vector_type(4)));
typedef float  v8f   __attribute__((ext_vector_type(8)));

#define TM 64
#define TN 64
#define TKS 32

// ---------------------------------------------------------------------------
// Generic GEMM: C[M,N] (epilogue) A[M,K](fp32 -> bf16) x Bw[K,N](bf16)
// mode 0: C = v + bias
// mode 1: C = gelu(v + bias)            (exact erf gelu)
// mode 3: C += rowscale[m] * (v + bias) (rowscale==nullptr -> scale 1.0)
// K must be a multiple of 32. Fast staging paths when M%64==0 / N%64==0.
// ---------------------------------------------------------------------------
__global__ __launch_bounds__(128) void gemm_bf16_wmma(
    const float* __restrict__ A, const bf16_t* __restrict__ Bw,
    const float* __restrict__ bias, float* __restrict__ C,
    const float* __restrict__ rowscale, int rs_stride,
    int M, int N, int K, int mode)
{
    // K-contiguous rows, 64B row stride -> aligned b128 fragment loads
    __shared__ bf16_t As [TM][TKS];
    __shared__ bf16_t BsT[TN][TKS];   // transposed B tile: BsT[n][k]

    const int tid  = threadIdx.x;
    const int lane = tid & 31;
    const int wave = tid >> 5;
    const int m0   = blockIdx.y * TM;
    const int n0   = blockIdx.x * TN;

    const bool fastA = ((M & (TM - 1)) == 0);
    const bool fastB = ((N & (TN - 1)) == 0);

    v8f acc[4];
    #pragma unroll
    for (int i = 0; i < 4; ++i)
        #pragma unroll
        for (int r = 0; r < 8; ++r) acc[i][r] = 0.0f;

    for (int k0 = 0; k0 < K; k0 += TKS) {
        // ---- stage A tile (64x32), fp32 -> bf16 ----
        if (fastA) {
            #pragma unroll
            for (int it = 0; it < 4; ++it) {
                const int base = it * 512 + tid * 4;       // 2048 elems / 4
                const int r = base >> 5, c = base & 31;
                const v4f v = *(const v4f*)(A + (size_t)(m0 + r) * K + (k0 + c));
                v4bf hv;
                hv[0] = (bf16_t)v[0]; hv[1] = (bf16_t)v[1];
                hv[2] = (bf16_t)v[2]; hv[3] = (bf16_t)v[3];
                *(v4bf*)&As[r][c] = hv;
            }
        } else {
            for (int i = tid; i < TM * TKS; i += 128) {
                const int r = i / TKS, c = i % TKS;
                const int gm = m0 + r;
                const float v = (gm < M) ? A[(size_t)gm * K + (k0 + c)] : 0.0f;
                As[r][c] = (bf16_t)v;
            }
        }
        // ---- stage B tile (32x64) transposed into BsT[n][k] ----
        if (fastB) {
            #pragma unroll
            for (int it = 0; it < 4; ++it) {
                const int base = it * 512 + tid * 4;       // base = k*64 + n
                const int k = base >> 6, n = base & 63;
                const v4bf hv = *(const v4bf*)(Bw + (size_t)(k0 + k) * N + (n0 + n));
                BsT[n + 0][k] = hv[0];
                BsT[n + 1][k] = hv[1];
                BsT[n + 2][k] = hv[2];
                BsT[n + 3][k] = hv[3];
            }
        } else {
            for (int i = tid; i < TKS * TN; i += 128) {
                const int k = i / TN, n = i % TN;
                const int gn = n0 + n;
                const bf16_t v = (gn < N) ? Bw[(size_t)(k0 + k) * N + gn] : (bf16_t)0.0f;
                BsT[n][k] = v;
            }
        }
        // prefetch next K-tile while this one computes (global_prefetch_b8)
        if (k0 + TKS < K) {
            __builtin_prefetch(A + (size_t)(m0 + (tid >> 1)) * K + (k0 + TKS), 0, 1);
            __builtin_prefetch(Bw + (size_t)(k0 + TKS + (tid & 31)) * N + n0, 0, 1);
        }
        __syncthreads();

        // ---- A fragment (16x32, ISA 7.12.2): two contiguous 8-half chunks ----
        const int ml  = wave * 16 + (lane & 15);
        const int kb2 = (lane < 16) ? 0 : 8;
        const v8bf a0 = *(const v8bf*)&As[ml][kb2];
        const v8bf a1 = *(const v8bf*)&As[ml][kb2 + 16];
        const v16bf afrag = __builtin_shufflevector(
            a0, a1, 0, 1, 2, 3, 4, 5, 6, 7, 8, 9, 10, 11, 12, 13, 14, 15);

        const int kbb = (lane < 16) ? 0 : 8;
        #pragma unroll
        for (int nt = 0; nt < 4; ++nt) {
            const int nl = nt * 16 + (lane & 15);
            const v8bf b0 = *(const v8bf*)&BsT[nl][kbb];
            const v8bf b1 = *(const v8bf*)&BsT[nl][kbb + 16];
            const v16bf bfrag = __builtin_shufflevector(
                b0, b1, 0, 1, 2, 3, 4, 5, 6, 7, 8, 9, 10, 11, 12, 13, 14, 15);
            acc[nt] = __builtin_amdgcn_wmma_f32_16x16x32_bf16(
                false, afrag, false, bfrag, (short)0, acc[nt], false, false);
        }
        __syncthreads();
    }

    // epilogue; C/D layout: VGPR r, lane -> n = lane%16, m = r + (lane<16?0:8)
    const int mbase = m0 + wave * 16 + ((lane < 16) ? 0 : 8);
    #pragma unroll
    for (int nt = 0; nt < 4; ++nt) {
        const int gn = n0 + nt * 16 + (lane & 15);
        if (gn >= N) continue;
        const float bv = bias ? bias[gn] : 0.0f;
        #pragma unroll
        for (int r = 0; r < 8; ++r) {
            const int gm = mbase + r;
            if (gm >= M) continue;
            float v = acc[nt][r] + bv;
            const size_t o = (size_t)gm * N + gn;
            if (mode == 1) {
                C[o] = 0.5f * v * (1.0f + erff(v * 0.70710678118654752440f));
            } else if (mode == 3) {
                const float s = rowscale ? rowscale[(size_t)gm * rs_stride] : 1.0f;
                C[o] += s * v;
            } else {
                C[o] = v;
            }
        }
    }
}

// ---------------------------------------------------------------------------
// weight fp32 -> bf16 (optionally transposed: dst[k,n] = src[n,k])
// ---------------------------------------------------------------------------
__global__ void wconv_k(const float* __restrict__ src, bf16_t* __restrict__ dst,
                        int K, int N, int transpose)
{
    const int i = blockIdx.x * 256 + threadIdx.x;
    if (i >= K * N) return;
    const int k = i / N, n = i % N;
    const float v = transpose ? src[(size_t)n * K + k] : src[i];
    dst[i] = (bf16_t)v;
}

// ---------------------------------------------------------------------------
// im2col for the patch-embed conv: out[b*196+p, c*256+py*16+px]
// ---------------------------------------------------------------------------
__global__ void im2col_k(const float* __restrict__ x, float* __restrict__ out)
{
    const int i = blockIdx.x * 256 + threadIdx.x;
    if (i >= 12544 * 768) return;
    const int r = i / 768, f = i % 768;
    const int b = r / 196, p = r % 196;
    const int gy = p / 14, gx = p % 14;
    const int c = f / 256, rem = f % 256;
    const int py = rem / 16, px = rem % 16;
    out[i] = x[(((size_t)b * 3 + c) * 224 + gy * 16 + py) * 224 + gx * 16 + px];
}

// ---------------------------------------------------------------------------
// z[b,s,:] = (s==0 ? cls : patch_out[b*196+s-1]) + pos[s]
// ---------------------------------------------------------------------------
__global__ void addclspos_k(const float* __restrict__ pe, const float* __restrict__ cls,
                            const float* __restrict__ pos, float* __restrict__ z)
{
    const int i = blockIdx.x * 256 + threadIdx.x;
    if (i >= 64 * 197 * 256) return;
    const int e = i & 255;
    const int t = i >> 8;
    const int s = t % 197;
    const int b = t / 197;
    const float v = (s == 0) ? cls[e] : pe[((size_t)b * 196 + (s - 1)) * 256 + e];
    z[i] = v + pos[s * 256 + e];
}

// ---------------------------------------------------------------------------
// LayerNorm over E=256 (one block per token)
// ---------------------------------------------------------------------------
__global__ __launch_bounds__(256) void layernorm_k(
    const float* __restrict__ in, const float* __restrict__ w,
    const float* __restrict__ b, float* __restrict__ out, int T)
{
    __shared__ float red[256];
    const int t = blockIdx.x;
    const int e = threadIdx.x;
    const float v = in[(size_t)t * 256 + e];
    red[e] = v;
    __syncthreads();
    for (int s = 128; s > 0; s >>= 1) { if (e < s) red[e] += red[e + s]; __syncthreads(); }
    const float mean = red[0] * (1.0f / 256.0f);
    __syncthreads();
    const float d = v - mean;
    red[e] = d * d;
    __syncthreads();
    for (int s = 128; s > 0; s >>= 1) { if (e < s) red[e] += red[e + s]; __syncthreads(); }
    const float var = red[0] * (1.0f / 256.0f);
    out[(size_t)t * 256 + e] = d * rsqrtf(var + 1e-5f) * w[e] + b[e];
}

// ---------------------------------------------------------------------------
// Attention: one block per (batch, head); online softmax, K/V staged in LDS
// qkv layout [T,768]: q=[0,256), k=[256,512), v=[512,768), head slice h*32
// ---------------------------------------------------------------------------
__global__ __launch_bounds__(256) void attn_k(const float* __restrict__ qkv,
                                              float* __restrict__ ctx)
{
    __shared__ float Ks[197][32];
    __shared__ float Vs[197][32];
    const int bh = blockIdx.x;
    const int b = bh >> 3;
    const int hd = bh & 7;
    const int tid = threadIdx.x;

    for (int i = tid; i < 197 * 32; i += 256) {
        const int s = i >> 5, d = i & 31;
        const size_t base = ((size_t)b * 197 + s) * 768 + hd * 32 + d;
        Ks[s][d] = qkv[base + 256];
        Vs[s][d] = qkv[base + 512];
    }
    __syncthreads();

    if (tid < 197) {
        const int s = tid;
        float q[32];
        const size_t qb = ((size_t)b * 197 + s) * 768 + hd * 32;
        #pragma unroll
        for (int d = 0; d < 32; ++d) q[d] = qkv[qb + d];

        const float scale = 0.17677669529663687f; // 1/sqrt(32)
        float m = -3.4e38f, den = 0.0f, acc[32];
        #pragma unroll
        for (int d = 0; d < 32; ++d) acc[d] = 0.0f;

        for (int j = 0; j < 197; ++j) {
            float dot = 0.0f;
            #pragma unroll
            for (int d = 0; d < 32; ++d) dot += q[d] * Ks[j][d];
            const float sc = dot * scale;
            if (sc > m) {
                const float f = expf(m - sc);
                den *= f;
                #pragma unroll
                for (int d = 0; d < 32; ++d) acc[d] *= f;
                m = sc;
            }
            const float p = expf(sc - m);
            den += p;
            #pragma unroll
            for (int d = 0; d < 32; ++d) acc[d] += p * Vs[j][d];
        }
        const float inv = 1.0f / den;
        const size_t ob = ((size_t)b * 197 + s) * 256 + hd * 32;
        #pragma unroll
        for (int d = 0; d < 32; ++d) ctx[ob + d] = acc[d] * inv;
    }
}

// ---------------------------------------------------------------------------
// Router: softmax over NE=4 logits, top-2 (first index wins ties), renormalize
// ---------------------------------------------------------------------------
__global__ void router_k(const float* __restrict__ xf, const float* __restrict__ rw,
                         const float* __restrict__ rb, float* __restrict__ gate, int T)
{
    const int t = blockIdx.x * 128 + threadIdx.x;
    if (t >= T) return;
    float l[4];
    #pragma unroll
    for (int n = 0; n < 4; ++n) l[n] = rb[n];
    const float* xr = xf + (size_t)t * 256;
    for (int k = 0; k < 256; ++k) {
        const float xv = xr[k];
        #pragma unroll
        for (int n = 0; n < 4; ++n) l[n] += xv * rw[k * 4 + n];
    }
    float m = l[0];
    #pragma unroll
    for (int n = 1; n < 4; ++n) m = fmaxf(m, l[n]);
    float p[4];
    #pragma unroll
    for (int n = 0; n < 4; ++n) p[n] = expf(l[n] - m);
    int i1 = 0;
    for (int n = 1; n < 4; ++n) if (p[n] > p[i1]) i1 = n;
    int i2 = -1;
    for (int n = 0; n < 4; ++n) { if (n == i1) continue; if (i2 < 0 || p[n] > p[i2]) i2 = n; }
    const float sum2 = p[i1] + p[i2];
    #pragma unroll
    for (int n = 0; n < 4; ++n)
        gate[(size_t)t * 4 + n] = (n == i1 || n == i2) ? p[n] / sum2 : 0.0f;
}

// ---------------------------------------------------------------------------
// cls extraction: cls[b,:] = z[b,0,:]
// ---------------------------------------------------------------------------
__global__ void takecls_k(const float* __restrict__ z, float* __restrict__ cls)
{
    const int i = blockIdx.x * 256 + threadIdx.x;
    if (i >= 64 * 256) return;
    const int b = i >> 8, e = i & 255;
    cls[i] = z[((size_t)b * 197) * 256 + e];
}

// ---------------------------------------------------------------------------
// host launch
// ---------------------------------------------------------------------------
extern "C" void kernel_launch(void* const* d_in, const int* in_sizes, int n_in,
                              void* d_out, int out_size, void* d_ws, size_t ws_size,
                              hipStream_t stream)
{
    (void)in_sizes; (void)n_in; (void)out_size; (void)ws_size;

    const float* x         = (const float*)d_in[0];
    const float* conv_w    = (const float*)d_in[1];
    const float* conv_b    = (const float*)d_in[2];
    const float* cls_token = (const float*)d_in[3];
    const float* pos_embed = (const float*)d_in[4];
    const float* ln1_w     = (const float*)d_in[5];
    const float* ln1_b     = (const float*)d_in[6];
    const float* qkv_w     = (const float*)d_in[7];
    const float* qkv_b     = (const float*)d_in[8];
    const float* out_w     = (const float*)d_in[9];
    const float* out_b     = (const float*)d_in[10];
    const float* ln2_w     = (const float*)d_in[11];
    const float* ln2_b     = (const float*)d_in[12];
    const float* router_w  = (const float*)d_in[13];
    const float* router_b  = (const float*)d_in[14];
    const float* e_w1      = (const float*)d_in[15];
    const float* e_b1      = (const float*)d_in[16];
    const float* e_w2      = (const float*)d_in[17];
    const float* e_b2      = (const float*)d_in[18];
    const float* norm_w    = (const float*)d_in[19];
    const float* norm_b    = (const float*)d_in[20];
    const float* h1_w      = (const float*)d_in[21];
    const float* h1_b      = (const float*)d_in[22];
    const float* h2_w      = (const float*)d_in[23];
    const float* h2_b      = (const float*)d_in[24];

    const int T = 64 * 197; // 12608

    char* ws = (char*)d_ws;
    size_t off = 0;
    auto alloc = [&](size_t bytes) -> void* {
        void* p = ws + off;
        off = (off + bytes + 255) & ~(size_t)255;
        return p;
    };

    float*  im2col = (float*)alloc((size_t)12544 * 768 * 4);
    float*  z      = (float*)alloc((size_t)T * 256 * 4);
    float*  hbuf   = (float*)alloc((size_t)T * 256 * 4);
    float*  qkvbuf = (float*)alloc((size_t)T * 768 * 4);
    float*  ctxbuf = (float*)alloc((size_t)T * 256 * 4);   // also patch-embed out
    float*  xf     = (float*)alloc((size_t)T * 256 * 4);
    float*  he     = (float*)alloc((size_t)T * 1024 * 4);
    float*  gate   = (float*)alloc((size_t)T * 4 * 4);
    bf16_t* wbf    = (bf16_t*)alloc((size_t)1024 * 1024 * 2);
    float*  clsbuf = (float*)alloc((size_t)64 * 256 * 4);
    float*  clsln  = (float*)alloc((size_t)64 * 256 * 4);
    float*  h1out  = (float*)alloc((size_t)64 * 256 * 4);

    auto gemm = [&](const float* A, const bf16_t* Bw, const float* bias, float* C,
                    const float* rs, int rss, int M, int N, int K, int mode) {
        dim3 g((N + TN - 1) / TN, (M + TM - 1) / TM);
        gemm_bf16_wmma<<<g, 128, 0, stream>>>(A, Bw, bias, C, rs, rss, M, N, K, mode);
    };
    auto wconv = [&](const float* src, bf16_t* dst, int K, int N, int tr) {
        const int n = K * N;
        wconv_k<<<(n + 255) / 256, 256, 0, stream>>>(src, dst, K, N, tr);
    };

    // ---- patch embedding ----
    im2col_k<<<(12544 * 768 + 255) / 256, 256, 0, stream>>>(x, im2col);
    wconv(conv_w, wbf, 768, 256, 1); // conv_w[256,768] -> Bw[768,256]
    gemm(im2col, wbf, conv_b, ctxbuf, nullptr, 0, 12544, 256, 768, 0);
    addclspos_k<<<(64 * 197 * 256 + 255) / 256, 256, 0, stream>>>(ctxbuf, cls_token, pos_embed, z);

    // ---- transformer layers ----
    for (int l = 0; l < 4; ++l) {
        // attention
        layernorm_k<<<T, 256, 0, stream>>>(z, ln1_w + l * 256, ln1_b + l * 256, hbuf, T);
        wconv(qkv_w + (size_t)l * 256 * 768, wbf, 256, 768, 0);
        gemm(hbuf, wbf, qkv_b + l * 768, qkvbuf, nullptr, 0, T, 768, 256, 0);
        attn_k<<<64 * 8, 256, 0, stream>>>(qkvbuf, ctxbuf);
        wconv(out_w + (size_t)l * 256 * 256, wbf, 256, 256, 0);
        gemm(ctxbuf, wbf, out_b + l * 256, z, nullptr, 0, T, 256, 256, 3); // z += ctx@W + b

        // MoE FFN
        layernorm_k<<<T, 256, 0, stream>>>(z, ln2_w + l * 256, ln2_b + l * 256, xf, T);
        router_k<<<(T + 127) / 128, 128, 0, stream>>>(
            xf, router_w + (size_t)l * 256 * 4, router_b + l * 4, gate, T);
        for (int n = 0; n < 4; ++n) {
            const size_t ei = (size_t)l * 4 + n;
            wconv(e_w1 + ei * 256 * 1024, wbf, 256, 1024, 0);
            gemm(xf, wbf, e_b1 + ei * 1024, he, nullptr, 0, T, 1024, 256, 1); // gelu
            wconv(e_w2 + ei * 1024 * 256, wbf, 1024, 256, 0);
            gemm(he, wbf, e_b2 + ei * 256, z, gate + n, 4, T, 256, 1024, 3); // z += g*(...)
        }
    }

    // ---- head ----
    takecls_k<<<(64 * 256 + 255) / 256, 256, 0, stream>>>(z, clsbuf);
    layernorm_k<<<64, 256, 0, stream>>>(clsbuf, norm_w, norm_b, clsln, 64);
    wconv(h1_w, wbf, 256, 256, 0);
    gemm(clsln, wbf, h1_b, h1out, nullptr, 0, 64, 256, 256, 1); // gelu
    wconv(h2_w, wbf, 256, 38, 0);
    gemm(h1out, wbf, h2_b, (float*)d_out, nullptr, 0, 64, 38, 256, 0);
}
